// FusedSTAttention_13769665151061
// MI455X (gfx1250) — matrix operand order repaired and checked by reading, not compiled, not run on hardware
//
#include <hip/hip_runtime.h>

typedef unsigned short u16;
typedef unsigned int u32;
typedef __attribute__((ext_vector_type(16))) __bf16 v16bf;
typedef __attribute__((ext_vector_type(8))) float v8f;

#define NTOK 196
#define CDIM 768
#define C3   2304
#define HEADS 12
#define HD   64
#define TT   16
#define BB   8
#define MROWS 25088      // B*T*N
#define BTH  1536        // B*T*H
#define QSZ  19267584    // MROWS*CDIM elements (per q/k/v tensor)
#define NPAD 224         // keys padded to 7 blocks of 32

__device__ __forceinline__ u16 f2bf(float f) {
  union { float f; u32 u; } c; c.f = f;
  u32 u = c.u + 0x7FFFu + ((c.u >> 16) & 1u);   // round-to-nearest-even
  return (u16)(u >> 16);
}
__device__ __forceinline__ float bf2f(u16 h) {
  union { u32 u; float f; } c; c.u = ((u32)h) << 16;
  return c.f;
}
// low 32 bits of a flat pointer into the LDS aperture == LDS byte offset (ISA 10.2)
__device__ __forceinline__ u32 lds_off(const void* p) { return (u32)(size_t)p; }

// async DMA: 16 bytes global -> LDS per lane, tracked by ASYNCcnt
__device__ __forceinline__ void async_b128(u32 ldsoff, const void* g) {
  asm volatile("global_load_async_to_lds_b128 %0, %1, off"
               :: "v"(ldsoff), "v"(g) : "memory");
}
__device__ __forceinline__ void wait_async0() {
  asm volatile("s_wait_asynccnt 0x0" ::: "memory");
}

// A-matrix 16x32 bf16 fragment (ISA 7.12.2): lane m = lane&15, hi half shifts K by 8;
// VGPRs 0-3: K pairs {0..7}+8hi, VGPRs 4-7: K pairs {16..23}+8hi. rowp = dword ptr to row@K0.
__device__ __forceinline__ v16bf load_a_frag(const u32* rowp, int hi) {
  union { v16bf v; u32 u[8]; } f;
  const u32* p = rowp + 4*hi;
  f.u[0]=p[0]; f.u[1]=p[1]; f.u[2]=p[2]; f.u[3]=p[3];
  f.u[4]=p[8]; f.u[5]=p[9]; f.u[6]=p[10]; f.u[7]=p[11];
  return f.v;
}
// B-matrix 32x16 bf16 fragment: lane n = lane&15, same K order as A: {0..7}+8hi, {16..23}+8hi.
__device__ __forceinline__ v16bf load_b_frag(const u32* rowp, int hi) {
  union { v16bf v; u32 u[8]; } f;
  const u32* p = rowp + 4*hi;
  f.u[0]=p[0]; f.u[1]=p[1]; f.u[2]=p[2]; f.u[3]=p[3];
  f.u[4]=p[8]; f.u[5]=p[9]; f.u[6]=p[10]; f.u[7]=p[11];
  return f.v;
}
__device__ __forceinline__ v8f wmma_bf16(v16bf a, v16bf b, v8f c) {
  return __builtin_amdgcn_wmma_f32_16x16x32_bf16(false, a, false, b, (short)0, c, false, false);
}
__device__ __forceinline__ float hmax16(float x) {
#pragma unroll
  for (int m=1;m<16;m<<=1) x = fmaxf(x, __shfl_xor(x, m, 32));
  return x;
}
__device__ __forceinline__ float hsum16(float x) {
#pragma unroll
  for (int m=1;m<16;m<<=1) x += __shfl_xor(x, m, 32);
  return x;
}

// ---------------- kernel 1: x / weights fp32 -> bf16 ----------------
__global__ void cvt_inputs(const float* __restrict__ x, const float* __restrict__ wq,
                           const float* __restrict__ wp, u16* __restrict__ xb,
                           u16* __restrict__ wqb, u16* __restrict__ wpb) {
  int i = blockIdx.x * 256 + threadIdx.x;
  if (i < MROWS * CDIM) xb[i] = f2bf(x[i]);
  if (i < CDIM * C3)    wqb[i] = f2bf(wq[i]);
  if (i < CDIM * CDIM)  wpb[i] = f2bf(wp[i]);
}

// ---------------- kernel 2: QKV GEMM (M=25088, K=768, N=2304) ----------------
// 128x128 tile per block; 8 waves in 4(M)x2(N); each wave owns 32x64; K-step 32.
// A tile staged via async DMA (bf16, contiguous); B tile transposed by VALU gather.
__global__ __launch_bounds__(256) void qkv_gemm(const u16* __restrict__ Xb,
                                                const u16* __restrict__ Wb,
                                                u16* __restrict__ QKV) {
  __shared__ u16 a_lds[128 * 32];
  __shared__ u16 b_lds[128 * 32];
  const int tid = threadIdx.x;
  const int lane = tid & 31, w = tid >> 5;
  const int hi = lane >> 4, l16 = lane & 15;
  const int row0 = blockIdx.y * 128, n0 = blockIdx.x * 128;
  const int wrow = (w & 3) * 32, wcol = (w >> 2) * 64;
  const u32 a_base = lds_off(a_lds);
  v8f acc[2][4];
#pragma unroll
  for (int i=0;i<2;i++)
#pragma unroll
    for (int j=0;j<4;j++)
#pragma unroll
      for (int v=0;v<8;v++) acc[i][j][v] = 0.f;

  for (int kt = 0; kt < 24; ++kt) {
    const int k0 = kt * 32;
    { // A tile: 128 rows x 64B, 512 x 16B chunks via async DMA
#pragma unroll
      for (int r=0;r<2;r++) {
        int c = tid + 256 * r;                 // chunk id: row = c>>2, part = c&3
        const void* g = (const void*)(Xb + (size_t)(row0 + (c >> 2)) * CDIM + k0 + (c & 3) * 8);
        async_b128(a_base + (u32)c * 16u, g);
      }
    }
    { // B tile: gather transposed, b_lds[n][k], n=0..127
      int n = tid & 127, g = tid >> 7;         // g in {0,1}: 16 K-rows each
      const u16* src = Wb + (size_t)(k0 + g * 16) * C3 + n0 + n;
      u16* dst = b_lds + n * 32 + g * 16;
#pragma unroll
      for (int i=0;i<16;i++) dst[i] = src[(size_t)i * C3];
    }
    if (kt < 23) {  // hint next K tiles toward the caches
      __builtin_prefetch((const void*)(Xb + (size_t)(row0 + (tid >> 1)) * CDIM + k0 + 32), 0, 0);
      __builtin_prefetch((const void*)(Wb + (size_t)(k0 + 32 + (tid >> 4)) * C3 + n0), 0, 0);
    }
    wait_async0();
    __syncthreads();
    v16bf aF[2], bF[4];
#pragma unroll
    for (int i=0;i<2;i++)
      aF[i] = load_a_frag((const u32*)(a_lds + (wrow + i*16 + l16) * 32), hi);
#pragma unroll
    for (int j=0;j<4;j++)
      bF[j] = load_b_frag((const u32*)(b_lds + (wcol + j*16 + l16) * 32), hi);
#pragma unroll
    for (int i=0;i<2;i++)
#pragma unroll
      for (int j=0;j<4;j++)
        acc[i][j] = wmma_bf16(aF[i], bF[j], acc[i][j]);
    __syncthreads();
  }
  // scatter: col -> (which, h, d); row -> (bt, n)
#pragma unroll
  for (int i=0;i<2;i++)
#pragma unroll
    for (int j=0;j<4;j++) {
      int colbase = n0 + wcol + j * 16;        // 16-col tile lies inside one head block
      int which = colbase / CDIM;
      int rem = colbase - which * CDIM;
      int h = rem / HD;
      int d = (rem % HD) + l16;
      u16* base = QKV + (size_t)which * QSZ;
#pragma unroll
      for (int v=0;v<8;v++) {
        int grow = row0 + wrow + i*16 + v + 8*hi;
        int ntok = grow % NTOK;
        int bt   = grow / NTOK;
        base[(((size_t)bt * HEADS + h) * NTOK + ntok) * HD + d] = f2bf(acc[i][j][v]);
      }
    }
}

// ---------------- kernel 3: spatial attention, flash-style ----------------
// one block per (bt,h); 4 waves; keys padded to 224; online softmax per 16-row tile.
// Q/K tiles are contiguous [196][64] bf16 -> staged via async DMA.
__global__ __launch_bounds__(128) void spatial_attn(const u16* __restrict__ QKV,
                                                    u16* __restrict__ X1) {
  extern __shared__ u16 smem[];
  u16* k_s  = smem;                   // [224][64]
  u16* q_s  = smem + NPAD * HD;       // [224][64]
  u16* vT_s = smem + 2 * NPAD * HD;   // [64][224]  (V transposed for B-frag loads)
  u16* p_s  = smem + 3 * NPAD * HD;   // [4][16*32] per-wave P staging (A layout)
  const int tid = threadIdx.x, lane = tid & 31, w = tid >> 5;
  const int hi = lane >> 4, l16 = lane & 15;
  const int bth = blockIdx.x;
  const int h = bth % HEADS;
  const int bt = bth / HEADS;
  const size_t base = (size_t)bth * (NTOK * HD);
  const u16* Q = QKV;
  const u16* K = QKV + (size_t)QSZ;
  const u16* V = QKV + (size_t)2 * QSZ;

  // async-stage Q and K (196*64 bf16 = 1568 x 16B chunks each)
  {
    const u32 qoff = lds_off(q_s), koff = lds_off(k_s);
    const char* qg = (const char*)(Q + base);
    const char* kg = (const char*)(K + base);
    for (int c = tid; c < 1568; c += 128) {
      async_b128(qoff + (u32)c * 16u, qg + (size_t)c * 16);
      async_b128(koff + (u32)c * 16u, kg + (size_t)c * 16);
    }
    // zero the 28 pad rows (196..223), 896 dwords per tensor
    u32* qd = (u32*)q_s; u32* kd = (u32*)k_s;
    for (int idx = tid; idx < (NPAD - NTOK) * 32; idx += 128) {
      qd[NTOK * 32 + idx] = 0u;
      kd[NTOK * 32 + idx] = 0u;
    }
  }
  // V transposed gather (element-wise)
  for (int idx = tid; idx < HD * NPAD; idx += 128) {
    int d = idx / NPAD, s = idx - d * NPAD;
    vT_s[idx] = (s < NTOK) ? V[base + (size_t)s * HD + d] : (u16)0;
  }
  wait_async0();
  __syncthreads();

  u16* pw = p_s + w * (16 * 32);
  for (int rt = w; rt < 13; rt += 4) {
    const u32* qrow = (const u32*)(q_s + (rt * 16 + l16) * HD);
    v16bf qA0 = load_a_frag(qrow, hi);
    v16bf qA1 = load_a_frag(qrow + 16, hi);
    float mrun[8], lrun[8];
    v8f o[4];
#pragma unroll
    for (int v=0;v<8;v++) { mrun[v] = -1e30f; lrun[v] = 0.f; }
#pragma unroll
    for (int c=0;c<4;c++)
#pragma unroll
      for (int v=0;v<8;v++) o[c][v] = 0.f;

    for (int j = 0; j < 7; ++j) {             // 7 key blocks of 32
      v8f S0, S1;
#pragma unroll
      for (int v=0;v<8;v++) { S0[v] = 0.f; S1[v] = 0.f; }
      const u32* kr0 = (const u32*)(k_s + (j * 32 + l16) * HD);
      const u32* kr1 = (const u32*)(k_s + (j * 32 + 16 + l16) * HD);
      S0 = wmma_bf16(qA0, load_b_frag(kr0, hi), S0);
      S0 = wmma_bf16(qA1, load_b_frag(kr0 + 16, hi), S0);
      S1 = wmma_bf16(qA0, load_b_frag(kr1, hi), S1);
      S1 = wmma_bf16(qA1, load_b_frag(kr1 + 16, hi), S1);

      const int col0 = j * 32 + l16;
      const bool ok0 = col0 < NTOK, ok1 = (col0 + 16) < NTOK;
      float corr[8];
#pragma unroll
      for (int v=0;v<8;v++) {
        float a = ok0 ? S0[v] * 0.125f : -1e30f;
        float b = ok1 ? S1[v] * 0.125f : -1e30f;
        float t = hmax16(fmaxf(a, b));
        float mnew = fmaxf(mrun[v], t);
        float p0 = __expf(a - mnew), p1 = __expf(b - mnew);
        float rs = hsum16(p0 + p1);
        corr[v] = __expf(mrun[v] - mnew);
        lrun[v] = lrun[v] * corr[v] + rs;
        mrun[v] = mnew;
        int pr = (v + 8 * hi) * 32;           // C-layout -> A-layout via LDS
        pw[pr + l16] = f2bf(p0);
        pw[pr + 16 + l16] = f2bf(p1);
      }
#pragma unroll
      for (int c=0;c<4;c++)
#pragma unroll
        for (int v=0;v<8;v++) o[c][v] *= corr[v];
      v16bf pA = load_a_frag((const u32*)(pw + l16 * 32), hi);
#pragma unroll
      for (int c=0;c<4;c++) {
        v16bf vB = load_b_frag((const u32*)(vT_s + (c * 16 + l16) * NPAD) + j * 16, hi);
        o[c] = wmma_bf16(pA, vB, o[c]);
      }
    }
#pragma unroll
    for (int v=0;v<8;v++) {
      int rowtok = rt * 16 + v + 8 * hi;
      if (rowtok < NTOK) {
        float inv = 1.f / lrun[v];
        size_t orow = ((size_t)bt * NTOK + rowtok) * CDIM + h * HD;
#pragma unroll
        for (int c=0;c<4;c++)
          X1[orow + c * 16 + l16] = f2bf(o[c][v] * inv);
      }
    }
  }
}

// ---------------- kernel 4: temporal attention (T=16: single tile) ----------------
__global__ __launch_bounds__(128) void temporal_attn(const u16* __restrict__ QKV,
                                                     u16* __restrict__ X2) {
  __shared__ u16 q_s[4][TT * HD];
  __shared__ u16 k_s[4][TT * HD];
  __shared__ u16 vT_s[4][HD * 32];   // [d][s], s padded 16->32 with zeros
  __shared__ u16 p_s[4][16 * 32];
  const int tid = threadIdx.x, lane = tid & 31, w = tid >> 5;
  const int hi = lane >> 4, l16 = lane & 15;
  const int g = blockIdx.x * 4 + w;          // token-head index
  const int n = g % NTOK;
  const int bh = g / NTOK;
  const int b = bh / HEADS, h = bh % HEADS;
  const u16* Q = QKV;
  const u16* K = QKV + (size_t)QSZ;
  const u16* V = QKV + (size_t)2 * QSZ;
  u32* qd = (u32*)q_s[w]; u32* kd = (u32*)k_s[w];
#pragma unroll
  for (int t=0;t<TT;t++) {
    size_t base = ((((size_t)(b * TT + t)) * HEADS + h) * NTOK + n) * HD;
    qd[t * 32 + lane] = ((const u32*)(Q + base))[lane];
    kd[t * 32 + lane] = ((const u32*)(K + base))[lane];
  }
#pragma unroll
  for (int r=0;r<2;r++) {
    int d = lane + r * 32;
#pragma unroll
    for (int t=0;t<TT;t++) {
      size_t base = ((((size_t)(b * TT + t)) * HEADS + h) * NTOK + n) * HD;
      vT_s[w][d * 32 + t] = V[base + d];
      vT_s[w][d * 32 + 16 + t] = 0;
    }
  }
  v16bf qA0 = load_a_frag((const u32*)(q_s[w] + l16 * HD), hi);
  v16bf qA1 = load_a_frag((const u32*)(q_s[w] + l16 * HD) + 16, hi);
  v16bf kB0 = load_b_frag((const u32*)(k_s[w] + l16 * HD), hi);
  v16bf kB1 = load_b_frag((const u32*)(k_s[w] + l16 * HD) + 16, hi);
  v8f S;
#pragma unroll
  for (int v=0;v<8;v++) S[v] = 0.f;
  S = wmma_bf16(qA0, kB0, S);
  S = wmma_bf16(qA1, kB1, S);
  float lrun[8];
#pragma unroll
  for (int v=0;v<8;v++) {
    float a = S[v] * 0.125f;
    float mx = hmax16(a);
    float p = __expf(a - mx);
    lrun[v] = hsum16(p);
    int pr = (v + 8 * hi) * 32;
    p_s[w][pr + l16] = f2bf(p);
    p_s[w][pr + 16 + l16] = 0;               // zero-pad K 16->32
  }
  v16bf pA = load_a_frag((const u32*)(p_s[w] + l16 * 32), hi);
#pragma unroll
  for (int c=0;c<4;c++) {
    v8f o;
#pragma unroll
    for (int v=0;v<8;v++) o[v] = 0.f;
    v16bf vB = load_b_frag((const u32*)(vT_s[w] + (c * 16 + l16) * 32), hi);
    o = wmma_bf16(pA, vB, o);
#pragma unroll
    for (int v=0;v<8;v++) {
      int t = v + 8 * hi;
      size_t orow = (((size_t)b * TT + t) * NTOK + n) * CDIM + h * HD;
      X2[orow + c * 16 + l16] = f2bf(o[v] / lrun[v]);
    }
  }
}

// ---------------- kernel 5: output projection, fused branch average + bias ----------------
__global__ __launch_bounds__(256) void proj_gemm(const u16* __restrict__ X1,
                                                 const u16* __restrict__ X2,
                                                 const u16* __restrict__ Wb,
                                                 const float* __restrict__ bias,
                                                 float* __restrict__ out) {
  __shared__ u16 a_lds[128 * 32];
  __shared__ u16 b_lds[128 * 32];
  const int tid = threadIdx.x, lane = tid & 31, w = tid >> 5;
  const int hi = lane >> 4, l16 = lane & 15;
  const int row0 = blockIdx.y * 128, n0 = blockIdx.x * 128;
  const int wrow = (w & 3) * 32, wcol = (w >> 2) * 64;
  v8f acc[2][4];
#pragma unroll
  for (int i=0;i<2;i++)
#pragma unroll
    for (int j=0;j<4;j++)
#pragma unroll
      for (int v=0;v<8;v++) acc[i][j][v] = 0.f;

  for (int kt = 0; kt < 24; ++kt) {
    const int k0 = kt * 32;
    { // A tile: 0.5*(x1+x2) -> bf16
      int m = tid >> 1, ks = (tid & 1) * 16;
      const u16* s1 = X1 + (size_t)(row0 + m) * CDIM + k0 + ks;
      const u16* s2 = X2 + (size_t)(row0 + m) * CDIM + k0 + ks;
      u16* dst = a_lds + m * 32 + ks;
#pragma unroll
      for (int i=0;i<16;i++) dst[i] = f2bf(0.5f * (bf2f(s1[i]) + bf2f(s2[i])));
    }
    { // B tile transposed
      int nn = tid & 127, gg = tid >> 7;
      const u16* src = Wb + (size_t)(k0 + gg * 16) * CDIM + n0 + nn;
      u16* dst = b_lds + nn * 32 + gg * 16;
#pragma unroll
      for (int i=0;i<16;i++) dst[i] = src[(size_t)i * CDIM];
    }
    if (kt < 23) {
      __builtin_prefetch((const void*)(X1 + (size_t)(row0 + (tid >> 1)) * CDIM + k0 + 32), 0, 0);
      __builtin_prefetch((const void*)(X2 + (size_t)(row0 + (tid >> 1)) * CDIM + k0 + 32), 0, 0);
    }
    __syncthreads();
    v16bf aF[2], bF[4];
#pragma unroll
    for (int i=0;i<2;i++)
      aF[i] = load_a_frag((const u32*)(a_lds + (wrow + i*16 + l16) * 32), hi);
#pragma unroll
    for (int j=0;j<4;j++)
      bF[j] = load_b_frag((const u32*)(b_lds + (wcol + j*16 + l16) * 32), hi);
#pragma unroll
    for (int i=0;i<2;i++)
#pragma unroll
      for (int j=0;j<4;j++)
        acc[i][j] = wmma_bf16(aF[i], bF[j], acc[i][j]);
    __syncthreads();
  }
#pragma unroll
  for (int i=0;i<2;i++)
#pragma unroll
    for (int j=0;j<4;j++) {
      int col = n0 + wcol + j * 16 + l16;
      float bc = bias[col];
#pragma unroll
      for (int v=0;v<8;v++) {
        int grow = row0 + wrow + i * 16 + v + 8 * hi;
        out[(size_t)grow * CDIM + col] = acc[i][j][v] + bc;
      }
    }
}

extern "C" void kernel_launch(void* const* d_in, const int* in_sizes, int n_in,
                              void* d_out, int out_size, void* d_ws, size_t ws_size,
                              hipStream_t stream) {
  const float* x  = (const float*)d_in[0];
  const float* wq = (const float*)d_in[1];
  const float* wp = (const float*)d_in[2];
  const float* bp = (const float*)d_in[3];
  float* out = (float*)d_out;

  // workspace (bf16 units): Xbf | QKV | X1 | X2 | Wqkv_bf | Wproj_bf (~236 MB)
  u16* Xbf = (u16*)d_ws;
  u16* QKV = Xbf + (size_t)QSZ;
  u16* X1  = QKV + (size_t)3 * QSZ;
  u16* X2  = X1 + (size_t)QSZ;
  u16* Wqb = X2 + (size_t)QSZ;
  u16* Wpb = Wqb + (size_t)CDIM * C3;

  cvt_inputs<<<(MROWS * CDIM + 255) / 256, 256, 0, stream>>>(x, wq, wp, Xbf, Wqb, Wpb);
  qkv_gemm<<<dim3(C3 / 128, MROWS / 128), 256, 0, stream>>>(Xbf, Wqb, QKV);
  size_t sp_lds = (size_t)(3 * NPAD * HD + 4 * 16 * 32) * sizeof(u16);  // 90112 B
  spatial_attn<<<BTH, 128, sp_lds, stream>>>(QKV, X1);
  temporal_attn<<<(BB * HEADS * NTOK) / 4, 128, 0, stream>>>(QKV, X2);
  proj_gemm<<<dim3(CDIM / 128, MROWS / 128), 256, 0, stream>>>(X1, X2, Wpb, bp, out);
}
